// DeformableConv2d_39719857553995
// MI455X (gfx1250) — matrix operand order repaired and checked by reading, not compiled, not run on hardware
//
#include <hip/hip_runtime.h>

// ---------------------------------------------------------------------------
// Deformable Conv2d (B=4, 64x64, C=256, 3x3, F=256) for gfx1250 (MI455X).
// Phase 1: bilinear gather -> bf16 A-tile (32 x 2304) in LDS.
// Phase 2: v_wmma_f32_16x16x32_bf16 GEMM; per K-chunk each wave does
//          2 A-frags (LDS) x 2 B-frags (global, pre-swizzled) = 4 WMMAs.
// ---------------------------------------------------------------------------

typedef __attribute__((ext_vector_type(16))) __bf16        v16bf;
typedef __attribute__((ext_vector_type(8)))  float         v8f;
typedef __attribute__((ext_vector_type(8)))  unsigned int  v8u;

#define BB    4
#define IHW   64
#define CC    256
#define FF    256
#define NN    9
#define KK    (NN * CC)      // 2304
#define KCH   32             // K per WMMA
#define NCH   (KK / KCH)     // 72 K-chunks
#define MT    32             // positions per workgroup
#define ROWS  2306           // padded A-tile row length (bf16); odd dword stride
#define TILE_B (MT * ROWS * 2)          // 147584 bytes
#define NCOORD (MT * NN)                // 288 (pos, tap) pairs
#define NFT   (FF / 16)                 // 16 feature tiles
#define WT_ELEMS (NFT * NCH * 32 * 16)  // 589824 bf16 values

// Reference's grid init after its (2,3,3)->(9,2) reshape quirk:
__constant__ int INIT0[NN] = {0, 0, 1, 1, 2, 1, 0, 2, 1};
__constant__ int INIT1[NN] = {0, 1, 1, 2, 0, 2, 1, 0, 2};

static __device__ __forceinline__ unsigned short f2bf_bits(float f) {
  unsigned u = __builtin_bit_cast(unsigned, f);
  return (unsigned short)((u + 0x7FFFu + ((u >> 16) & 1u)) >> 16);
}

// --------------------------------------------------------------------------
// Prep: W (N,C,F) f32  ->  d_ws bf16 in exact WMMA B-fragment per-lane order:
//   idx = ((ftile*NCH + chunk)*32 + lane)*16 + e
//   f = ftile*16 + (lane&15),  K = chunk*32 + (e>>3)*16 + (lane>>4)*8 + (e&7)
// A wave's B fragment is one contiguous 1KB block (32B/lane).
// --------------------------------------------------------------------------
__global__ __launch_bounds__(256) void deform_prep_w(
    const float* __restrict__ W, unsigned short* __restrict__ wt) {
  int tid = blockIdx.x * 256 + threadIdx.x;
  if (tid >= WT_ELEMS) return;
  int e     = tid & 15;
  int lane  = (tid >> 4) & 31;
  int rest  = tid >> 9;            // ftile*NCH + chunk
  int ch    = rest % NCH;
  int ftile = rest / NCH;
  int f     = ftile * 16 + (lane & 15);
  int K     = ch * KCH + (e >> 3) * 16 + (lane >> 4) * 8 + (e & 7);
  wt[tid] = f2bf_bits(W[K * FF + f]);
}

// --------------------------------------------------------------------------
// Main kernel: one workgroup = 32 output positions, all 256 features.
// --------------------------------------------------------------------------
__global__ __launch_bounds__(256) void deform_wmma_kernel(
    const float* __restrict__ x,            // (B,64,64,256)
    const float* __restrict__ offs,         // (B,64,64,18)
    const unsigned short* __restrict__ wt,  // pre-swizzled bf16 W fragments
    const float* __restrict__ bias,         // (256,)
    float* __restrict__ out) {              // (B,64,64,256)
  extern __shared__ char smem[];
  unsigned short* sTile = (unsigned short*)smem;                 // MT x ROWS bf16
  int*   sAddr = (int*)  (smem + TILE_B);                        // [4][288]
  float* sWgt  = (float*)(smem + TILE_B + 4 * NCOORD * 4);       // [4][288]

  const int t     = threadIdx.x;
  const int mbase = blockIdx.x * MT;

  // ---- coords: 32 positions x 9 taps ----
  for (int ci = t; ci < NCOORD; ci += 256) {
    int p = ci / NN, n = ci - p * NN;
    int m = mbase + p;
    int b = m >> 12;
    int h = (m >> 6) & 63;
    int w = m & 63;
    float cy = (float)(h - 1 + INIT0[n]) + offs[m * (2 * NN) + 2 * n];
    float cx = (float)(w - 1 + INIT1[n]) + offs[m * (2 * NN) + 2 * n + 1];
    cy = fminf(fmaxf(cy, 0.0f), 63.0f);
    cx = fminf(fmaxf(cx, 0.0f), 63.0f);
    float fy0 = floorf(cy), fx0 = floorf(cx);
    int y0 = (int)fy0, x0 = (int)fx0;
    int y1 = (int)ceilf(cy), x1 = (int)ceilf(cx);
    float fy = cy - fy0, fx = cx - fx0;
    int bb = b << 20;                      // b * 64*64*256
    sAddr[0 * NCOORD + ci] = bb + (y0 << 14) + (x0 << 8);   // (y0,x0)
    sAddr[1 * NCOORD + ci] = bb + (y0 << 14) + (x1 << 8);   // (y0,x1)
    sAddr[2 * NCOORD + ci] = bb + (y1 << 14) + (x0 << 8);   // (y1,x0)
    sAddr[3 * NCOORD + ci] = bb + (y1 << 14) + (x1 << 8);   // (y1,x1)
    sWgt[0 * NCOORD + ci] = (1.0f - fy) * (1.0f - fx);
    sWgt[1 * NCOORD + ci] = (1.0f - fy) * fx;
    sWgt[2 * NCOORD + ci] = fy * (1.0f - fx);
    sWgt[3 * NCOORD + ci] = fy * fx;
  }
  __syncthreads();

  // ---- gather 32 x 2304 samples -> bf16 A-tile (c == thread id: coalesced) ----
  for (int i = 0; i < (MT * KK) / 256; ++i) {   // 288 iterations
    int v   = i * 256 + t;
    int pos = v / KK;
    int k   = v - pos * KK;
    int idx = pos * NN + (k >> 8);
    int c   = k & 255;
    float val = sWgt[idx]              * x[sAddr[idx] + c]
              + sWgt[NCOORD + idx]     * x[sAddr[NCOORD + idx] + c]
              + sWgt[2 * NCOORD + idx] * x[sAddr[2 * NCOORD + idx] + c]
              + sWgt[3 * NCOORD + idx] * x[sAddr[3 * NCOORD + idx] + c];
    sTile[pos * ROWS + k] = f2bf_bits(val);
  }
  __syncthreads();

  // ---- WMMA GEMM: 8 waves, each: 2 row-blocks x 2 feature tiles ----
  const int lane = t & 31;
  const int wv   = t >> 5;
  const int half = lane >> 4;       // K-half selector (A) / row-half (D)
  const int r    = lane & 15;       // A row / D column
  const unsigned int* arow0 = (const unsigned int*)(smem + r * (ROWS * 2));
  const unsigned int* arow1 = (const unsigned int*)(smem + (16 + r) * (ROWS * 2));
  const int ft0 = wv * 2;
  const int ft1 = wv * 2 + 1;
  const v8u* bp0 = (const v8u*)wt + (size_t)(ft0 * NCH) * 32 + lane;
  const v8u* bp1 = (const v8u*)wt + (size_t)(ft1 * NCH) * 32 + lane;

  v8f acc00 = {};   // rows 0-15,  ftile0
  v8f acc10 = {};   // rows 16-31, ftile0
  v8f acc01 = {};   // rows 0-15,  ftile1
  v8f acc11 = {};   // rows 16-31, ftile1

  for (int ch = 0; ch < NCH; ++ch) {
    // A fragments (ISA 16-bit A layout): dword j -> K = (j/4)*16 + half*8 + (2j&7)
    int basd = ch * 16 + half * 4;
    v8u au0, au1;
#pragma unroll
    for (int j = 0; j < 4; ++j) {
      au0[j]     = arow0[basd + j];
      au0[j + 4] = arow0[basd + 8 + j];
      au1[j]     = arow1[basd + j];
      au1[j + 4] = arow1[basd + 8 + j];
    }
    v16bf af0 = __builtin_bit_cast(v16bf, au0);
    v16bf af1 = __builtin_bit_cast(v16bf, au1);
    // B fragments: one contiguous 32B load per lane (pre-swizzled)
    v16bf bf0 = __builtin_bit_cast(v16bf, bp0[ch * 32]);
    v16bf bf1 = __builtin_bit_cast(v16bf, bp1[ch * 32]);
    acc00 = __builtin_amdgcn_wmma_f32_16x16x32_bf16(false, af0, false, bf0,
                                                    (short)0, acc00, false, false);
    acc10 = __builtin_amdgcn_wmma_f32_16x16x32_bf16(false, af1, false, bf0,
                                                    (short)0, acc10, false, false);
    acc01 = __builtin_amdgcn_wmma_f32_16x16x32_bf16(false, af0, false, bf1,
                                                    (short)0, acc01, false, false);
    acc11 = __builtin_amdgcn_wmma_f32_16x16x32_bf16(false, af1, false, bf1,
                                                    (short)0, acc11, false, false);
  }

  // ---- epilogue: D layout => VGPR v -> M = v + 8*half, N = lane&15 ----
  {
    float bv0 = bias[ft0 * 16 + r];
    float bv1 = bias[ft1 * 16 + r];
    int rowA = mbase + half * 8;        // rows 0-15 block
    int rowB = mbase + 16 + half * 8;   // rows 16-31 block
#pragma unroll
    for (int vv = 0; vv < 8; ++vv) {
      out[(size_t)(rowA + vv) * FF + ft0 * 16 + r] = acc00[vv] + bv0;
      out[(size_t)(rowB + vv) * FF + ft0 * 16 + r] = acc10[vv] + bv0;
      out[(size_t)(rowA + vv) * FF + ft1 * 16 + r] = acc01[vv] + bv1;
      out[(size_t)(rowB + vv) * FF + ft1 * 16 + r] = acc11[vv] + bv1;
    }
  }
}

// --------------------------------------------------------------------------
extern "C" void kernel_launch(void* const* d_in, const int* in_sizes, int n_in,
                              void* d_out, int out_size, void* d_ws, size_t ws_size,
                              hipStream_t stream) {
  const float* x    = (const float*)d_in[0];
  const float* offs = (const float*)d_in[1];
  const float* W    = (const float*)d_in[2];
  const float* bias = (const float*)d_in[3];
  float* out = (float*)d_out;
  unsigned short* wt = (unsigned short*)d_ws;  // needs 589824*2 = 1.18 MB scratch

  // 1) swizzle + bf16-convert weights into B-fragment order
  deform_prep_w<<<(WT_ELEMS + 255) / 256, 256, 0, stream>>>(W, wt);

  // 2) gather + WMMA GEMM
  size_t smem_bytes = TILE_B + 2 * 4 * NCOORD * 4;   // 147584 + 9216 = 156800
  int nblocks = (BB * IHW * IHW) / MT;               // 512
  deform_wmma_kernel<<<nblocks, 256, smem_bytes, stream>>>(x, offs, wt, bias, out);
}
